// MistralAttention_59279138619549
// MI455X (gfx1250) — matrix operand order repaired and checked by reading, not compiled, not run on hardware
//
#include <hip/hip_runtime.h>

// ---------------------------------------------------------------------------
// Mistral sliding-window attention block for MI455X (gfx1250, wave32, WMMA).
// B=1, S=4096, HID=1024, H=8, D=128, KVH=2, WINDOW=2048, theta=1e4.
// All matmuls run on v_wmma_f32_16x16x32_bf16 (bf16 in, fp32 accumulate).
// ---------------------------------------------------------------------------

#define S_LEN  4096
#define HIDDEN 1024
#define NH     8
#define NKVH   2
#define HD     128
#define KVW    (NKVH * HD)   // 256
#define WINDOW 2048

typedef __attribute__((ext_vector_type(16))) __bf16 v16bf;
typedef __attribute__((ext_vector_type(8)))  float  v8f;

union Frag {
    v16bf v;
    uint4 q[2];
    unsigned int u[8];
};

__device__ __forceinline__ unsigned short f2bf(float f) {
    unsigned int u = __float_as_uint(f);
    u += 0x7FFFu + ((u >> 16) & 1u);          // round-to-nearest-even
    return (unsigned short)(u >> 16);
}
__device__ __forceinline__ float bf2f(unsigned short h) {
    return __uint_as_float(((unsigned int)h) << 16);
}
// XOR-16 lane swap inside a wave32 (ds_swizzle group-of-32: and=0x1f, xor=0x10)
__device__ __forceinline__ float xor16(float x) {
    return __int_as_float(__builtin_amdgcn_ds_swizzle(__float_as_int(x), 0x401F));
}
__device__ __forceinline__ unsigned int pack2bf(float a, float b) {
    return (unsigned int)f2bf(a) | ((unsigned int)f2bf(b) << 16);
}

// ---------------------------------------------------------------------------
// fp32 -> bf16 cast
// ---------------------------------------------------------------------------
__global__ void cast_f32_bf16(const float* __restrict__ src,
                              unsigned short* __restrict__ dst, int n) {
    for (int i = blockIdx.x * blockDim.x + threadIdx.x; i < n;
         i += gridDim.x * blockDim.x)
        dst[i] = f2bf(src[i]);
}

// ---------------------------------------------------------------------------
// C[M,N] = A[M,K] @ B[N,K]^T  (bf16 operands, f32 accumulate).
// Block = 128 threads = 4 waves; each wave computes a 16(M) x 64(N) strip.
// Grid = (M/64, N/64). All tile addresses are one base + constant offsets.
// ---------------------------------------------------------------------------
template <bool OUTF32>
__global__ void gemm_bt(const unsigned short* __restrict__ A,
                        const unsigned short* __restrict__ B,
                        void* __restrict__ Cout, int M, int N, int K) {
    const int lane = threadIdx.x & 31;
    // wave-uniform wave id -> scalar addressing/loop control
    const int wid  = __builtin_amdgcn_readfirstlane(threadIdx.x >> 5);
    const int mb   = blockIdx.x * 64 + wid * 16;
    const int nb   = blockIdx.y * 64;
    const bool lo  = lane < 16;
    const int mr   = lane & 15;
    const int sel8 = lo ? 0 : 8;

    v8f zero = {0.f, 0.f, 0.f, 0.f, 0.f, 0.f, 0.f, 0.f};
    v8f acc[4];
#pragma unroll
    for (int t = 0; t < 4; ++t) acc[t] = zero;

    const unsigned short* arow = A + (size_t)(mb + mr) * K + sel8;
    const unsigned short* brow = B + (size_t)(nb + mr) * K + sel8;
    for (int kb = 0; kb < K; kb += 32) {
        Frag af;  // A-operand: lane<16 holds k 0..7 & 16..23, hi lanes 8..15 & 24..31
        af.q[0] = *(const uint4*)(arow + kb);
        af.q[1] = *(const uint4*)(arow + kb + 16);
#pragma unroll
        for (int t = 0; t < 4; ++t) {
            Frag bf;  // B-operand: lane<16 holds k 0..7 & 16..23 of col n, hi lanes 8..15 & 24..31
            const unsigned short* p = brow + (size_t)t * (16 * 1024) + kb;
            bf.q[0] = *(const uint4*)(p);
            bf.q[1] = *(const uint4*)(p + 16);
            acc[t] = __builtin_amdgcn_wmma_f32_16x16x32_bf16(
                false, af.v, false, bf.v, (short)0, acc[t], false, false);
        }
    }
    const int mrow0 = mb + sel8;
#pragma unroll
    for (int t = 0; t < 4; ++t) {
#pragma unroll
        for (int i = 0; i < 8; ++i) {
            size_t idx = (size_t)(mrow0 + i) * N + nb + t * 16 + mr;
            if (OUTF32) ((float*)Cout)[idx] = acc[t][i];
            else        ((unsigned short*)Cout)[idx] = f2bf(acc[t][i]);
        }
    }
}

// ---------------------------------------------------------------------------
// In-place RoPE on a [S, nheads*128] bf16 tensor. One thread per (s, head, j<64).
// ---------------------------------------------------------------------------
__global__ void rope_kernel(unsigned short* __restrict__ T,
                            const int* __restrict__ pos, int nheads) {
    int idx = blockIdx.x * blockDim.x + threadIdx.x;
    int total = S_LEN * nheads * 64;
    if (idx >= total) return;
    int j  = idx & 63;
    int t2 = idx >> 6;
    int hh = t2 % nheads;
    int s  = t2 / nheads;
    float p    = (float)pos[s];
    float invf = __expf(-(float)j * 0.14391156831212787f);  // ln(1e4)/64
    float fr   = p * invf;
    float sn, cs;
    __sincosf(fr, &sn, &cs);
    unsigned short* base = T + (size_t)s * (nheads * HD) + hh * HD + j;
    float x1 = bf2f(base[0]);
    float x2 = bf2f(base[64]);
    base[0]  = f2bf(x1 * cs - x2 * sn);
    base[64] = f2bf(x2 * cs + x1 * sn);
}

// ---------------------------------------------------------------------------
// Vt[c, s] = V[s, c]   (bf16, C = 256 columns)
// ---------------------------------------------------------------------------
__global__ void transpose_bf16(const unsigned short* __restrict__ V,
                               unsigned short* __restrict__ Vt) {
    int idx = blockIdx.x * blockDim.x + threadIdx.x;
    if (idx >= S_LEN * KVW) return;
    int c = idx % KVW;
    int s = idx / KVW;
    Vt[(size_t)c * S_LEN + s] = V[idx];
}

// ---------------------------------------------------------------------------
// Flash attention, GQA, sliding-window causal mask.
// Grid (S/64, NH); 4 waves/block; each wave owns 16 queries of one head.
// S^T = K x Q^T (query lives in a lane; softmax reductions = one XOR-16
// swizzle), then O^T = V^T x P^T. Interior chunks skip all mask logic.
// launch_bounds(128,1): free the full VGPR budget (no spills, deeper
// load/wmma pipelining).
// ---------------------------------------------------------------------------
__global__ void __launch_bounds__(128, 1)
attn_kernel(const unsigned short* __restrict__ Q,
            const unsigned short* __restrict__ Kb,
            const unsigned short* __restrict__ Vt,
            unsigned short* __restrict__ ctx) {
    const int lane = threadIdx.x & 31;
    // wave-uniform wave id: makes qb/kb/full scalar (s_cmp/s_cbranch loop)
    const int wid  = __builtin_amdgcn_readfirstlane(threadIdx.x >> 5);
    const int qb   = blockIdx.x * 64 + wid * 16;
    const int h    = blockIdx.y;
    const int kvh  = h >> 2;  // repeat_interleave: head h uses kv head h/4
    const bool lo  = lane < 16;
    const int mr   = lane & 15;
    const int qrow = qb + mr;
    const int sel8 = lo ? 0 : 8;
    const float scale = 0.08838834764831845f;  // 1/sqrt(128)

    // Q fragments (B-operand, 4 chunks of 32 head-dims)
    Frag qf[4];
    const unsigned short* qp = Q + (size_t)qrow * HIDDEN + h * HD + sel8;
#pragma unroll
    for (int c = 0; c < 4; ++c) {
        qf[c].q[0] = *(const uint4*)(qp + c * 32);
        qf[c].q[1] = *(const uint4*)(qp + c * 32 + 16);
    }

    v8f zero = {0.f, 0.f, 0.f, 0.f, 0.f, 0.f, 0.f, 0.f};
    v8f accO[8];
#pragma unroll
    for (int t = 0; t < 8; ++t) accO[t] = zero;
    float m = -__builtin_inff();
    float l = 0.f;

    // loop-invariant bases; per-iteration address = base + kb * stride + const
    const unsigned short* kbase = Kb + (size_t)mr * KVW + kvh * HD + sel8;
    const unsigned short* vbase = Vt + (size_t)(kvh * HD + mr) * S_LEN + (lo ? 0 : 16);

    int kstart = qb - (WINDOW - 1);
    if (kstart < 0) kstart = 0;
    kstart &= ~31;
    const int kend = qb + 15;

    for (int kb = kstart; kb <= kend; kb += 32) {
        const unsigned short* kp = kbase + (size_t)kb * KVW;
        // ---- S^T tiles: (keys kb..kb+15) and (kb+16..kb+31) x 16 queries ----
        v8f s0 = zero, s1 = zero;
#pragma unroll
        for (int c = 0; c < 4; ++c) {
            Frag kf;
            kf.q[0] = *(const uint4*)(kp + c * 32);
            kf.q[1] = *(const uint4*)(kp + c * 32 + 16);
            s0 = __builtin_amdgcn_wmma_f32_16x16x32_bf16(
                false, kf.v, false, qf[c].v, (short)0, s0, false, false);
        }
#pragma unroll
        for (int c = 0; c < 4; ++c) {
            Frag kf;
            kf.q[0] = *(const uint4*)(kp + 16 * KVW + c * 32);
            kf.q[1] = *(const uint4*)(kp + 16 * KVW + c * 32 + 16);
            s1 = __builtin_amdgcn_wmma_f32_16x16x32_bf16(
                false, kf.v, false, qf[c].v, (short)0, s1, false, false);
        }

        // chunk fully valid for every query of this wave? (scalar branch)
        const bool full = (kb + 31 <= qb) && (kb >= qb + 16 - WINDOW);

        float mnew, alpha, ps = 0.f;
        Frag pf;  // P^T B-operand (32 keys x 16 queries)

        if (full) {
            // ---- interior fast path: no masking ----
            float mloc = s0[0];
#pragma unroll
            for (int i = 1; i < 8; ++i) mloc = fmaxf(mloc, s0[i]);
#pragma unroll
            for (int i = 0; i < 8; ++i) mloc = fmaxf(mloc, s1[i]);
            mloc *= scale;
            float mx = fmaxf(mloc, xor16(mloc));
            mnew  = fmaxf(m, mx);
            alpha = __expf(m - mnew);  // m==-inf -> 0, mnew finite here
#pragma unroll
            for (int j = 0; j < 4; ++j) {
                float a0 = __expf(s0[2 * j] * scale - mnew);
                float a1 = __expf(s0[2 * j + 1] * scale - mnew);
                float b0 = __expf(s1[2 * j] * scale - mnew);
                float b1 = __expf(s1[2 * j + 1] * scale - mnew);
                ps += (a0 + a1) + (b0 + b1);
                float ra0 = xor16(a0), ra1 = xor16(a1);
                float rb0 = xor16(b0), rb1 = xor16(b1);
                pf.u[j]     = pack2bf(lo ? a0 : rb0, lo ? a1 : rb1);
                pf.u[4 + j] = pack2bf(lo ? ra0 : b0, lo ? ra1 : b1);
            }
        } else {
            // ---- boundary path: per-element causal + window mask ----
            float mloc = -__builtin_inff();
#pragma unroll
            for (int i = 0; i < 8; ++i) {
                int k0 = kb + i + sel8;
                int k1 = k0 + 16;
                if (k0 <= qrow && k0 > qrow - WINDOW)
                    mloc = fmaxf(mloc, s0[i] * scale);
                if (k1 <= qrow && k1 > qrow - WINDOW)
                    mloc = fmaxf(mloc, s1[i] * scale);
            }
            float mx = fmaxf(mloc, xor16(mloc));
            mnew = fmaxf(m, mx);
            if      (mnew < -1e37f) alpha = 1.f;  // nothing valid seen yet
            else if (m    < -1e37f) alpha = 0.f;  // first chunk with valid keys
            else                    alpha = __expf(m - mnew);
#pragma unroll
            for (int j = 0; j < 4; ++j) {
                int k0 = kb + 2 * j + sel8;
                int k1 = k0 + 16;
                bool va0 = (k0     <= qrow) && (k0     > qrow - WINDOW);
                bool va1 = (k0 + 1 <= qrow) && (k0 + 1 > qrow - WINDOW);
                bool vb0 = (k1     <= qrow) && (k1     > qrow - WINDOW);
                bool vb1 = (k1 + 1 <= qrow) && (k1 + 1 > qrow - WINDOW);
                float a0 = va0 ? __expf(s0[2 * j] * scale - mnew) : 0.f;
                float a1 = va1 ? __expf(s0[2 * j + 1] * scale - mnew) : 0.f;
                float b0 = vb0 ? __expf(s1[2 * j] * scale - mnew) : 0.f;
                float b1 = vb1 ? __expf(s1[2 * j + 1] * scale - mnew) : 0.f;
                ps += (a0 + a1) + (b0 + b1);
                float ra0 = xor16(a0), ra1 = xor16(a1);
                float rb0 = xor16(b0), rb1 = xor16(b1);
                pf.u[j]     = pack2bf(lo ? a0 : rb0, lo ? a1 : rb1);
                pf.u[4 + j] = pack2bf(lo ? ra0 : b0, lo ? ra1 : b1);
            }
        }

        l = l * alpha + ps;
        m = mnew;
#pragma unroll
        for (int t = 0; t < 8; ++t)
#pragma unroll
            for (int i = 0; i < 8; ++i) accO[t][i] *= alpha;

        // ---- O^T += V^T x P^T (8 d-tiles; constant 128KB tile offsets) ----
        const unsigned short* vp = vbase + kb;
#pragma unroll
        for (int dt = 0; dt < 8; ++dt) {
            Frag vf;
            vf.q[0] = *(const uint4*)(vp + dt * (16 * S_LEN));
            vf.q[1] = *(const uint4*)(vp + dt * (16 * S_LEN) + 8);
            accO[dt] = __builtin_amdgcn_wmma_f32_16x16x32_bf16(
                false, vf.v, false, pf.v, (short)0, accO[dt], false, false);
        }
    }

    // ---- normalize and write ctx (bf16, [s, HID] layout, paired dword stores) ----
    float lt   = l + xor16(l);
    float invl = 1.f / lt;
    unsigned short* cp = ctx + (size_t)qrow * HIDDEN + h * HD + sel8;
#pragma unroll
    for (int dt = 0; dt < 8; ++dt) {
#pragma unroll
        for (int j = 0; j < 4; ++j) {
            unsigned int w = pack2bf(accO[dt][2 * j] * invl,
                                     accO[dt][2 * j + 1] * invl);
            *(unsigned int*)(cp + dt * 16 + 2 * j) = w;
        }
    }
}

// ---------------------------------------------------------------------------
extern "C" void kernel_launch(void* const* d_in, const int* in_sizes, int n_in,
                              void* d_out, int out_size, void* d_ws,
                              size_t ws_size, hipStream_t stream) {
    const float* X  = (const float*)d_in[0];
    const float* wq = (const float*)d_in[1];
    const float* wk = (const float*)d_in[2];
    const float* wv = (const float*)d_in[3];
    const float* wo = (const float*)d_in[4];
    const int*  pos = (const int*)d_in[5];

    unsigned short* ws = (unsigned short*)d_ws;
    size_t o = 0;
    unsigned short* Xb  = ws + o; o += (size_t)S_LEN * HIDDEN;
    unsigned short* Wqb = ws + o; o += (size_t)HIDDEN * HIDDEN;
    unsigned short* Wkb = ws + o; o += (size_t)KVW * HIDDEN;
    unsigned short* Wvb = ws + o; o += (size_t)KVW * HIDDEN;
    unsigned short* Wob = ws + o; o += (size_t)HIDDEN * HIDDEN;
    unsigned short* Qb  = ws + o; o += (size_t)S_LEN * HIDDEN;
    unsigned short* Kb  = ws + o; o += (size_t)S_LEN * KVW;
    unsigned short* Vb  = ws + o; o += (size_t)S_LEN * KVW;
    unsigned short* Vt  = ws + o; o += (size_t)S_LEN * KVW;
    unsigned short* Ctx = ws + o; o += (size_t)S_LEN * HIDDEN;
    // total ws use: ~35 MB (bf16)

    auto blocks = [](int n) { int g = (n + 255) / 256; return g < 4096 ? g : 4096; };

    // bf16 casts
    cast_f32_bf16<<<blocks(S_LEN * HIDDEN), 256, 0, stream>>>(X, Xb, S_LEN * HIDDEN);
    cast_f32_bf16<<<blocks(HIDDEN * HIDDEN), 256, 0, stream>>>(wq, Wqb, HIDDEN * HIDDEN);
    cast_f32_bf16<<<blocks(KVW * HIDDEN), 256, 0, stream>>>(wk, Wkb, KVW * HIDDEN);
    cast_f32_bf16<<<blocks(KVW * HIDDEN), 256, 0, stream>>>(wv, Wvb, KVW * HIDDEN);
    cast_f32_bf16<<<blocks(HIDDEN * HIDDEN), 256, 0, stream>>>(wo, Wob, HIDDEN * HIDDEN);

    // Q/K/V projections (WMMA)
    gemm_bt<false><<<dim3(S_LEN / 64, HIDDEN / 64), 128, 0, stream>>>(
        Xb, Wqb, Qb, S_LEN, HIDDEN, HIDDEN);
    gemm_bt<false><<<dim3(S_LEN / 64, KVW / 64), 128, 0, stream>>>(
        Xb, Wkb, Kb, S_LEN, KVW, HIDDEN);
    gemm_bt<false><<<dim3(S_LEN / 64, KVW / 64), 128, 0, stream>>>(
        Xb, Wvb, Vb, S_LEN, KVW, HIDDEN);

    // RoPE on Q and K
    rope_kernel<<<blocks(S_LEN * NH * 64), 256, 0, stream>>>(Qb, pos, NH);
    rope_kernel<<<blocks(S_LEN * NKVH * 64), 256, 0, stream>>>(Kb, pos, NKVH);

    // V transpose for the O^T = V^T x P^T formulation
    transpose_bf16<<<blocks(S_LEN * KVW), 256, 0, stream>>>(Vb, Vt);

    // flash attention
    attn_kernel<<<dim3(S_LEN / 64, NH), 128, 0, stream>>>(Qb, Kb, Vt, Ctx);

    // output projection (fp32 out)
    gemm_bt<true><<<dim3(S_LEN / 64, HIDDEN / 64), 128, 0, stream>>>(
        Ctx, Wob, (float*)d_out, S_LEN, HIDDEN, HIDDEN);
}
